// TransformerBlock_43636867728089
// MI455X (gfx1250) — matrix operand order repaired and checked by reading, not compiled, not run on hardware
//
#include <hip/hip_runtime.h>
#include <hip/hip_bf16.h>
#include <hip/hip_fp16.h>

typedef _Float16 f16;
typedef __attribute__((ext_vector_type(16))) _Float16 v16h;
typedef __attribute__((ext_vector_type(8)))  float    v8f;

union Frag { v16h h; uint4 q[2]; };
union H8   { f16 h[8]; uint4 q; };

#define P_TOT 32768      // B*H*W
#define DIMC  384
#define HW    4096

// ---------------------------------------------------------------------------
// CDNA5 async global->LDS copy (ASYNCcnt tracked). 16 bytes per lane.
// vdst VGPR holds the LDS byte offset, vaddr the 64-bit global address.
// ---------------------------------------------------------------------------
__device__ __forceinline__ unsigned lds_off32(const void* p) {
    return (unsigned)(unsigned long long)
        (__attribute__((address_space(3))) const void*)p;
}
#define ASYNC_B128(ldsoff, gptr)                                   \
    asm volatile("global_load_async_to_lds_b128 %0, %1, off"       \
                 :: "v"(ldsoff), "v"(gptr) : "memory")
#define WAIT_ASYNC(n)                                              \
    asm volatile("s_wait_asynccnt " #n ::: "memory")

// ---------------------------------------------------------------------------
// Weight pre-conversion f32 -> f16 with zero padding / split remap.
// Split semantics: indices < split map direct; [split,1024) are zero pad;
// >=1024 map back to (i - (1024-split)). split==0 -> simple tail pad.
// ---------------------------------------------------------------------------
__device__ inline int mapi(int i, int split, int S) {
    if (split == 0) return (i < S) ? i : -1;
    if (i < split) return i;
    if (i < 1024) return -1;
    int j = i - (1024 - split);
    return (j < S) ? j : -1;
}

__global__ void pad_w_k(const float* __restrict__ src, f16* __restrict__ dst,
                        int Nd, int Kd, int Ns, int Ks, int nsplit, int ksplit)
{
    size_t idx = (size_t)blockIdx.x * 256 + threadIdx.x;
    if (idx >= (size_t)Nd * Kd) return;
    int n = (int)(idx / Kd), k = (int)(idx % Kd);
    int sn = mapi(n, nsplit, Ns), sk = mapi(k, ksplit, Ks);
    dst[idx] = (sn >= 0 && sk >= 0) ? (f16)src[(size_t)sn * Ks + sk] : (f16)0.f;
}

__global__ void pad_b_k(const float* __restrict__ src, float* __restrict__ dst,
                        int Nd, int Ns, int nsplit)
{
    int n = blockIdx.x * 256 + threadIdx.x;
    if (n >= Nd) return;
    int sn = mapi(n, nsplit, Ns);
    dst[n] = (sn >= 0) ? src[sn] : 0.f;
}

// ---------------------------------------------------------------------------
// LayerNorm over channels (C=384). One wave per pixel, wave32 shuffles.
// Input NCHW f32 (template true) or NHWC f32 (false); output NHWC f16.
// ---------------------------------------------------------------------------
template<bool NCHW_IN>
__global__ __launch_bounds__(256)
void layernorm_k(const float* __restrict__ src,
                 const float* __restrict__ gam,
                 const float* __restrict__ bet,
                 f16* __restrict__ dst)
{
    int lane = threadIdx.x & 31;
    int wid  = threadIdx.x >> 5;
    int pix  = blockIdx.x * 8 + wid;
    int bb = pix >> 12, n = pix & 4095;
    float v[12];
#pragma unroll
    for (int j = 0; j < 12; ++j) {
        int ch = lane + 32 * j;
        v[j] = NCHW_IN ? src[((size_t)(bb * DIMC + ch)) * HW + n]
                       : src[(size_t)pix * DIMC + ch];
    }
    float s = 0.f, s2 = 0.f;
#pragma unroll
    for (int j = 0; j < 12; ++j) { s += v[j]; s2 += v[j] * v[j]; }
#pragma unroll
    for (int off = 16; off > 0; off >>= 1) {
        s  += __shfl_xor(s,  off);
        s2 += __shfl_xor(s2, off);
    }
    float mean = s * (1.0f / DIMC);
    float var  = s2 * (1.0f / DIMC) - mean * mean;
    float rstd = rsqrtf(var + 1e-5f);
#pragma unroll
    for (int j = 0; j < 12; ++j) {
        int ch = lane + 32 * j;
        dst[(size_t)pix * DIMC + ch] = (f16)((v[j] - mean) * rstd * gam[ch] + bet[ch]);
    }
}

// ---------------------------------------------------------------------------
// Generic WMMA GEMM:  C[pix][n] = sum_k act[pix][k] * wgt[n][k] + bias[n]
// WMMA roles: M = out-channel (from weights), N = pixel (from activations).
// Block: 256 threads = 8 waves; tile 64 ch x 128 pix; K step 32.
// Tiles stream global->LDS with CDNA5 async DMA, double buffered:
// issue tile kt+1, s_wait_asynccnt 3 (3 ops/tile/wave, in-order) -> tile kt
// resident while kt+1 is in flight.
// EPI: 0 = f16 store [pix*ldc+n]
//      1 = pout: + x(NCHW f32) residual -> f32 NHWC outf
//      2 = gelu(y)*y -> f16 store
//      3 = fout: + xattn(NHWC f32) residual -> f32 NCHW dout
// ---------------------------------------------------------------------------
template<int EPI>
__global__ __launch_bounds__(256)
void gemm_nt(const f16* __restrict__ act, int lda,
             const f16* __restrict__ wgt,
             const float* __restrict__ bias,
             int K, int ldc,
             f16* __restrict__ outh,
             float* __restrict__ outf,
             const float* __restrict__ extra,
             float* __restrict__ dout)
{
    __shared__ alignas(16) f16 lW[2][64][40];
    __shared__ alignas(16) f16 lA[2][128][40];

    const int tid  = threadIdx.x;
    const int lane = tid & 31;
    const int wid  = tid >> 5;
    const int wm   = wid & 1;        // ch half  (0..1) -> 32 ch
    const int wn   = wid >> 1;       // pix quad (0..3) -> 32 pix
    const int ch0  = blockIdx.x * 64;
    const int p0   = blockIdx.y * 128;
    const int m    = lane & 15;
    const int hi   = lane >> 4;

    // per-thread DMA slots: 1 x 16B of the W tile, 2 x 16B of the A tile
    const int rW  = tid >> 2,          cW  = (tid & 3) * 8;
    const int rA0 = tid >> 2,          cA0 = (tid & 3) * 8;
    const int rA1 = (tid + 256) >> 2,  cA1 = (tid & 3) * 8;

    auto issue = [&](int buf, int k0) {
        ASYNC_B128(lds_off32(&lW[buf][rW][cW]),
                   wgt + (size_t)(ch0 + rW) * K + k0 + cW);
        ASYNC_B128(lds_off32(&lA[buf][rA0][cA0]),
                   act + (size_t)(p0 + rA0) * lda + k0 + cA0);
        ASYNC_B128(lds_off32(&lA[buf][rA1][cA1]),
                   act + (size_t)(p0 + rA1) * lda + k0 + cA1);
    };

    v8f zero = {};
    v8f acc[2][2];
#pragma unroll
    for (int t = 0; t < 2; ++t)
#pragma unroll
        for (int s = 0; s < 2; ++s) acc[t][s] = zero;

    const int nk = K / 32;
    issue(0, 0);
    for (int kt = 0; kt < nk; ++kt) {
        const int cur = kt & 1;
        if (kt + 1 < nk) {
            issue(cur ^ 1, (kt + 1) * 32);
            WAIT_ASYNC(3);          // previous 3 (current tile) retired
        } else {
            WAIT_ASYNC(0);
        }
        __syncthreads();

        Frag fa[2], fb[2];
#pragma unroll
        for (int t = 0; t < 2; ++t) {
            const f16* row = &lW[cur][32 * wm + 16 * t + m][0];
            fa[t].q[0] = *(const uint4*)(row + hi * 8);
            fa[t].q[1] = *(const uint4*)(row + 16 + hi * 8);
        }
#pragma unroll
        for (int s = 0; s < 2; ++s) {
            const f16* row = &lA[cur][32 * wn + 16 * s + m][0];
            fb[s].q[0] = *(const uint4*)(row + hi * 8);
            fb[s].q[1] = *(const uint4*)(row + 16 + hi * 8);
        }
#pragma unroll
        for (int t = 0; t < 2; ++t)
#pragma unroll
            for (int s = 0; s < 2; ++s)
                acc[t][s] = __builtin_amdgcn_wmma_f32_16x16x32_f16(
                    false, fa[t].h, false, fb[s].h, (short)0, acc[t][s],
                    false, false);
        __syncthreads();            // reads done before buffer reuse
    }

#pragma unroll
    for (int t = 0; t < 2; ++t) {
#pragma unroll
        for (int s = 0; s < 2; ++s) {
            int pix = p0 + 32 * wn + 16 * s + m;
            int chb = ch0 + 32 * wm + 16 * t + 8 * hi;
            float v[8];
#pragma unroll
            for (int r = 0; r < 8; ++r) v[r] = acc[t][s][r] + bias[chb + r];
            if (EPI == 0) {
                H8 o;
#pragma unroll
                for (int r = 0; r < 8; ++r) o.h[r] = (f16)v[r];
                *(uint4*)(outh + (size_t)pix * ldc + chb) = o.q;
            } else if (EPI == 2) {
                H8 o;
#pragma unroll
                for (int r = 0; r < 8; ++r) {
                    float y = v[r];
                    float g = 0.5f * y * (1.0f + erff(y * 0.70710678118f));
                    o.h[r] = (f16)(g * y);
                }
                *(uint4*)(outh + (size_t)pix * ldc + chb) = o.q;
            } else if (EPI == 1) {
                int bb = pix >> 12, n = pix & 4095;
#pragma unroll
                for (int r = 0; r < 8; ++r)
                    v[r] += extra[((size_t)(bb * DIMC + chb + r)) * HW + n];
#pragma unroll
                for (int r = 0; r < 8; ++r)
                    outf[(size_t)pix * DIMC + chb + r] = v[r];
            } else { // EPI == 3
                int bb = pix >> 12, n = pix & 4095;
#pragma unroll
                for (int r = 0; r < 8; ++r)
                    dout[((size_t)(bb * DIMC + chb + r)) * HW + n] =
                        v[r] + extra[(size_t)pix * DIMC + chb + r];
            }
        }
    }
}

// ---------------------------------------------------------------------------
// Depthwise 3x3 conv on qkv (NHWC f16 in). Writes q,k -> NCHW f32, v -> NHWC f16.
// ---------------------------------------------------------------------------
__global__ void dwconv_qkv_k(const f16* __restrict__ qkv,
                             const float* __restrict__ w9,
                             const float* __restrict__ bias,
                             float* __restrict__ dwq,
                             float* __restrict__ dwk,
                             f16* __restrict__ vnhwc)
{
    size_t idx = (size_t)blockIdx.x * 256 + threadIdx.x;
    if (idx >= (size_t)P_TOT * 1152) return;
    int c = (int)(idx % 1152);
    int p = (int)(idx / 1152);
    int bb = p >> 12, n = p & 4095, hh = n >> 6, ww = n & 63;
    float acc = bias[c];
#pragma unroll
    for (int dy = -1; dy <= 1; ++dy) {
        int hy = hh + dy;
        if (hy < 0 || hy > 63) continue;
#pragma unroll
        for (int dx = -1; dx <= 1; ++dx) {
            int wx = ww + dx;
            if (wx < 0 || wx > 63) continue;
            acc += (float)qkv[((size_t)(bb * 4096 + hy * 64 + wx)) * 1152 + c]
                   * w9[c * 9 + (dy + 1) * 3 + (dx + 1)];
        }
    }
    if (c < 384)      dwq[((size_t)(bb * DIMC + c)) * HW + n] = acc;
    else if (c < 768) dwk[((size_t)(bb * DIMC + (c - 384))) * HW + n] = acc;
    else              vnhwc[(size_t)p * DIMC + (c - 768)] = (f16)acc;
}

// Depthwise 3x3 on x1 (cols [0,1021) of p_h, ld 2048) -> cat cols [0,1024)
__global__ void dwconv_hid_k(const f16* __restrict__ ph,
                             const float* __restrict__ w9,
                             const float* __restrict__ bias,
                             f16* __restrict__ cath)
{
    size_t idx = (size_t)blockIdx.x * 256 + threadIdx.x;
    if (idx >= (size_t)P_TOT * 1024) return;
    int c = (int)(idx % 1024);
    int p = (int)(idx / 1024);
    if (c >= 1021) { cath[(size_t)p * 2048 + c] = (f16)0.f; return; }
    int bb = p >> 12, n = p & 4095, hh = n >> 6, ww = n & 63;
    float acc = bias[c];
#pragma unroll
    for (int dy = -1; dy <= 1; ++dy) {
        int hy = hh + dy;
        if (hy < 0 || hy > 63) continue;
#pragma unroll
        for (int dx = -1; dx <= 1; ++dx) {
            int wx = ww + dx;
            if (wx < 0 || wx > 63) continue;
            acc += (float)ph[((size_t)(bb * 4096 + hy * 64 + wx)) * 2048 + c]
                   * w9[c * 9 + (dy + 1) * 3 + (dx + 1)];
        }
    }
    cath[(size_t)p * 2048 + c] = (f16)acc;
}

// ---------------------------------------------------------------------------
// spf l2-norm over W (192 rows of 64), then 3x3 conv 3->384 for spm.
// ---------------------------------------------------------------------------
__global__ void spf_norm_k(const float* __restrict__ spf, float* __restrict__ spfn)
{
    int r = threadIdx.x;
    if (r >= 192) return;
    const float* src = spf + r * 64;
    float s = 0.f;
    for (int i = 0; i < 64; ++i) { float x = src[i]; s += x * x; }
    float inv = 1.0f / fmaxf(sqrtf(s), 1e-12f);
    for (int i = 0; i < 64; ++i) spfn[r * 64 + i] = src[i] * inv;
}

__global__ void spm_k(const float* __restrict__ spfn,
                      const float* __restrict__ pw,
                      const float* __restrict__ pb,
                      float* __restrict__ spm)
{
    int idx = blockIdx.x * 256 + threadIdx.x;
    if (idx >= DIMC * HW) return;
    int c = idx >> 12, n = idx & 4095, hh = n >> 6, ww = n & 63;
    float acc = pb[c];
    for (int i = 0; i < 3; ++i)
        for (int dy = -1; dy <= 1; ++dy) {
            int hy = hh + dy;
            if (hy < 0 || hy > 63) continue;
            for (int dx = -1; dx <= 1; ++dx) {
                int wx = ww + dx;
                if (wx < 0 || wx > 63) continue;
                acc += spfn[i * HW + hy * 64 + wx]
                       * pw[((c * 3 + i) * 3 + dy + 1) * 3 + dx + 1];
            }
        }
    spm[idx] = acc;
}

// ---------------------------------------------------------------------------
// q path: per (b,c) block -- row l2n over W(64), *spm, global l2n over N(4096).
// ---------------------------------------------------------------------------
__global__ __launch_bounds__(256)
void qnorm_k(const float* __restrict__ dwq, const float* __restrict__ spm,
             f16* __restrict__ qs)
{
    __shared__ float red[256];
    int bc = blockIdx.x;              // b*384 + c
    int c = bc % DIMC;
    const float* src = dwq + (size_t)bc * HW;
    int tid = threadIdx.x;
    int hh = tid >> 2, q4 = (tid & 3) * 16;
    float v[16];
    float rs = 0.f;
#pragma unroll
    for (int i = 0; i < 16; ++i) { v[i] = src[hh * 64 + q4 + i]; rs += v[i] * v[i]; }
    red[tid] = rs;
    __syncthreads();
    float rowsq = red[hh * 4] + red[hh * 4 + 1] + red[hh * 4 + 2] + red[hh * 4 + 3];
    float rinv = 1.0f / fmaxf(sqrtf(rowsq), 1e-12f);
    float s2 = 0.f;
#pragma unroll
    for (int i = 0; i < 16; ++i) {
        v[i] = (v[i] * rinv) * spm[(size_t)c * HW + hh * 64 + q4 + i];
        s2 += v[i] * v[i];
    }
    __syncthreads();
    red[tid] = s2;
    __syncthreads();
    for (int s = 128; s > 0; s >>= 1) {
        if (tid < s) red[tid] += red[tid + s];
        __syncthreads();
    }
    float inv = 1.0f / fmaxf(sqrtf(red[0]), 1e-12f);
#pragma unroll
    for (int i = 0; i < 16; ++i)
        qs[(size_t)bc * HW + hh * 64 + q4 + i] = (f16)(v[i] * inv);
}

__global__ __launch_bounds__(256)
void knorm_k(const float* __restrict__ dwk, f16* __restrict__ kr)
{
    __shared__ float red[256];
    int bc = blockIdx.x;
    const float* src = dwk + (size_t)bc * HW;
    int tid = threadIdx.x;
    float v[16];
    float s2 = 0.f;
#pragma unroll
    for (int i = 0; i < 16; ++i) { v[i] = src[tid * 16 + i]; s2 += v[i] * v[i]; }
    red[tid] = s2;
    __syncthreads();
    for (int s = 128; s > 0; s >>= 1) {
        if (tid < s) red[tid] += red[tid + s];
        __syncthreads();
    }
    float inv = 1.0f / fmaxf(sqrtf(red[0]), 1e-12f);
#pragma unroll
    for (int i = 0; i < 16; ++i)
        kr[(size_t)bc * HW + tid * 16 + i] = (f16)(v[i] * inv);
}

// ---------------------------------------------------------------------------
// attn = qs . kr^T (48x48, K=4096) per (b,h) + topk-softmax combine.
// Produces combined blend matrix Mcomb (48x64, zero padded) in f16:
//   Mcomb = w0*(a1+a2+a3+a4)*relu(attn) + w1*sum_i c_i * masked_softmax_ki(attn)
// K split across 8 waves; deterministic sequential LDS reduction.
// ---------------------------------------------------------------------------
__global__ __launch_bounds__(256)
void attn_k(const f16* __restrict__ qs, const f16* __restrict__ kr,
            const float* __restrict__ temperature,
            const float* __restrict__ a1, const float* __restrict__ a2,
            const float* __restrict__ a3, const float* __restrict__ a4,
            const float* __restrict__ wmix,
            f16* __restrict__ attnM)
{
    __shared__ float sA[48][48];
    int tid = threadIdx.x, lane = tid & 31, wid = tid >> 5;
    int bh = blockIdx.x, bb = bh >> 3, h = bh & 7;
    for (int i = tid; i < 48 * 48; i += 256) ((float*)sA)[i] = 0.f;
    __syncthreads();

    const f16* qb = qs + ((size_t)(bb * DIMC + h * 48)) * HW;
    const f16* kb = kr + ((size_t)(bb * DIMC + h * 48)) * HW;
    int m = lane & 15, hi = lane >> 4;
    v8f zero = {};
    v8f acc[3][3];
#pragma unroll
    for (int t = 0; t < 3; ++t)
#pragma unroll
        for (int s = 0; s < 3; ++s) acc[t][s] = zero;

    for (int k0 = wid * 512; k0 < (wid + 1) * 512; k0 += 32) {
        Frag fa[3], fb[3];
#pragma unroll
        for (int t = 0; t < 3; ++t) {
            const f16* row = qb + (size_t)(16 * t + m) * HW + k0;
            fa[t].q[0] = *(const uint4*)(row + hi * 8);
            fa[t].q[1] = *(const uint4*)(row + 16 + hi * 8);
        }
#pragma unroll
        for (int s = 0; s < 3; ++s) {
            const f16* row = kb + (size_t)(16 * s + m) * HW + k0;
            fb[s].q[0] = *(const uint4*)(row + hi * 8);
            fb[s].q[1] = *(const uint4*)(row + 16 + hi * 8);
        }
#pragma unroll
        for (int t = 0; t < 3; ++t)
#pragma unroll
            for (int s = 0; s < 3; ++s)
                acc[t][s] = __builtin_amdgcn_wmma_f32_16x16x32_f16(
                    false, fa[t].h, false, fb[s].h, (short)0, acc[t][s],
                    false, false);
    }
    for (int w = 0; w < 8; ++w) {
        if (wid == w) {
#pragma unroll
            for (int t = 0; t < 3; ++t)
#pragma unroll
                for (int s = 0; s < 3; ++s)
#pragma unroll
                    for (int r = 0; r < 8; ++r)
                        sA[16 * t + 8 * hi + r][16 * s + m] += acc[t][s][r];
        }
        __syncthreads();
    }

    if (tid < 48) {
        int c = tid;
        float temp = temperature[h];
        float aa[48];
        for (int d = 0; d < 48; ++d) aa[d] = sA[c][d] * temp;
        float mw = fmaxf(wmix[0], wmix[1]);
        float e0 = expf(wmix[0] - mw), e1 = expf(wmix[1] - mw);
        float w0 = e0 / (e0 + e1), w1 = e1 / (e0 + e1);
        float S = a1[0] + a2[0] + a3[0] + a4[0];
        float Mrow[48];
        for (int d = 0; d < 48; ++d) Mrow[d] = w0 * S * fmaxf(aa[d], 0.f);
        const int kks[4] = {24, 32, 36, 38};
        float thr[4] = {0.f, 0.f, 0.f, 0.f};
        for (int d = 0; d < 48; ++d) {
            int rk = 0;
            for (int j = 0; j < 48; ++j)
                rk += (aa[j] > aa[d]) || (aa[j] == aa[d] && j < d);
            for (int i = 0; i < 4; ++i)
                if (rk == kks[i] - 1) thr[i] = aa[d];
        }
        const float* cs[4] = {a1, a2, a3, a4};
        for (int i = 0; i < 4; ++i) {
            float ci = cs[i][0];
            float mx = -1e30f;
            for (int d = 0; d < 48; ++d) {
                float md = (aa[d] >= thr[i]) ? aa[d] : aa[d] * 1e-6f;
                mx = fmaxf(mx, md);
            }
            float Z = 0.f;
            for (int d = 0; d < 48; ++d) {
                float md = (aa[d] >= thr[i]) ? aa[d] : aa[d] * 1e-6f;
                Z += expf(md - mx);
            }
            float sc = w1 * ci / Z;
            for (int d = 0; d < 48; ++d) {
                float md = (aa[d] >= thr[i]) ? aa[d] : aa[d] * 1e-6f;
                Mrow[d] += expf(md - mx) * sc;
            }
        }
        f16* dst = attnM + ((size_t)bh * 48 + c) * 64;
        for (int d = 0; d < 48; ++d) dst[d] = (f16)Mrow[d];
        for (int d = 48; d < 64; ++d) dst[d] = (f16)0.f;
    }
}

// ---------------------------------------------------------------------------
// out = Mcomb(48x64) @ v (K padded to 64; pad cols of Mcomb are zero so the
// over-read of v rows 48..63 contributes nothing). Output NHWC f16.
// ---------------------------------------------------------------------------
__global__ __launch_bounds__(256)
void outv_k(const f16* __restrict__ attnM, const f16* __restrict__ vnhwc,
            f16* __restrict__ outh)
{
    int tid = threadIdx.x, lane = tid & 31, wid = tid >> 5;
    int h = blockIdx.y, bb = blockIdx.z;
    int pixb = blockIdx.x * 128 + wid * 16;
    const f16* Mb = attnM + (size_t)(bb * 8 + h) * 48 * 64;
    int m = lane & 15, hi = lane >> 4;
    v8f zero = {};
    v8f acc[3] = {zero, zero, zero};
#pragma unroll
    for (int k0 = 0; k0 < 64; k0 += 32) {
        Frag fa[3], fb;
#pragma unroll
        for (int t = 0; t < 3; ++t) {
            const f16* row = Mb + (16 * t + m) * 64 + k0;
            fa[t].q[0] = *(const uint4*)(row + hi * 8);
            fa[t].q[1] = *(const uint4*)(row + 16 + hi * 8);
        }
        const f16* vrow = vnhwc + ((size_t)(bb * 4096 + pixb + m)) * DIMC
                          + h * 48 + k0;
        fb.q[0] = *(const uint4*)(vrow + hi * 8);
        fb.q[1] = *(const uint4*)(vrow + 16 + hi * 8);
#pragma unroll
        for (int t = 0; t < 3; ++t)
            acc[t] = __builtin_amdgcn_wmma_f32_16x16x32_f16(
                false, fa[t].h, false, fb.h, (short)0, acc[t], false, false);
    }
    int pix = pixb + m;
#pragma unroll
    for (int t = 0; t < 3; ++t) {
        int chb = h * 48 + 16 * t + 8 * hi;
        H8 o;
#pragma unroll
        for (int r = 0; r < 8; ++r) o.h[r] = (f16)acc[t][r];
        *(uint4*)(outh + ((size_t)(bb * 4096 + pix)) * DIMC + chb) = o.q;
    }
}

// ---------------------------------------------------------------------------
extern "C" void kernel_launch(void* const* d_in, const int* in_sizes, int n_in,
                              void* d_out, int out_size, void* d_ws, size_t ws_size,
                              hipStream_t stream)
{
    (void)in_sizes; (void)n_in; (void)out_size; (void)ws_size;
    const float* x       = (const float*)d_in[0];
    const float* spf     = (const float*)d_in[1];
    const float* ln1_w   = (const float*)d_in[2];
    const float* ln1_b   = (const float*)d_in[3];
    const float* qkv_w   = (const float*)d_in[4];
    const float* qkv_b   = (const float*)d_in[5];
    const float* qkvdw_w = (const float*)d_in[6];
    const float* qkvdw_b = (const float*)d_in[7];
    const float* proj_w  = (const float*)d_in[8];
    const float* proj_b  = (const float*)d_in[9];
    const float* temper  = (const float*)d_in[10];
    const float* a1      = (const float*)d_in[11];
    const float* a2      = (const float*)d_in[12];
    const float* a3      = (const float*)d_in[13];
    const float* a4      = (const float*)d_in[14];
    const float* wmix    = (const float*)d_in[15];
    const float* pout_w  = (const float*)d_in[16];
    const float* pout_b  = (const float*)d_in[17];
    const float* ln2_w   = (const float*)d_in[18];
    const float* ln2_b   = (const float*)d_in[19];
    const float* pin_w   = (const float*)d_in[20];
    const float* pin_b   = (const float*)d_in[21];
    const float* dw_w    = (const float*)d_in[22];
    const float* dw_b    = (const float*)d_in[23];
    const float* lin_w   = (const float*)d_in[24];
    const float* lin_b   = (const float*)d_in[25];
    const float* adj_w   = (const float*)d_in[26];
    const float* adj_b   = (const float*)d_in[27];
    const float* fout_w  = (const float*)d_in[28];
    const float* fout_b  = (const float*)d_in[29];
    float* dout = (float*)d_out;

    char* ws = (char*)d_ws;
    size_t off = 0;
    auto alloc = [&](size_t bytes) -> void* {
        void* p = ws + off;
        off += (bytes + 255) & ~(size_t)255;
        return p;
    };
    // persistent region (weights f16 + padded biases + attention residual)
    f16* qkv_wh  = (f16*)alloc((size_t)1152 * 384 * 2);
    f16* pout_wh = (f16*)alloc((size_t)384 * 384 * 2);
    f16* pin_wh  = (f16*)alloc((size_t)2048 * 384 * 2);
    f16* lin_wh  = (f16*)alloc((size_t)1024 * 1024 * 2);
    f16* adj_wh  = (f16*)alloc((size_t)1024 * 2048 * 2);
    f16* fout_wh = (f16*)alloc((size_t)384 * 1024 * 2);
    float* pin_bp = (float*)alloc(2048 * 4);
    float* lin_bp = (float*)alloc(1024 * 4);
    float* adj_bp = (float*)alloc(1024 * 4);
    float* xattn  = (float*)alloc((size_t)P_TOT * DIMC * 4);
    size_t mark = off;
    // phase 1 (attention)
    f16*   xn1  = (f16*)alloc((size_t)P_TOT * DIMC * 2);
    f16*   qkvh = (f16*)alloc((size_t)P_TOT * 1152 * 2);
    float* dwq  = (float*)alloc((size_t)P_TOT * DIMC * 4);
    float* dwk  = (float*)alloc((size_t)P_TOT * DIMC * 4);
    f16*   vnh  = (f16*)alloc((size_t)P_TOT * DIMC * 2 + 1024);
    float* spfn = (float*)alloc((size_t)3 * HW * 4);
    float* spmb = (float*)alloc((size_t)DIMC * HW * 4);
    f16*   qsb  = (f16*)alloc((size_t)P_TOT * DIMC * 2);
    f16*   krb  = (f16*)alloc((size_t)P_TOT * DIMC * 2);
    f16*   atM  = (f16*)alloc((size_t)64 * 48 * 64 * 2);
    f16*   outh = (f16*)alloc((size_t)P_TOT * DIMC * 2);
    // phase 2 (FFN) aliases phase 1
    off = mark;
    f16* xn2  = (f16*)alloc((size_t)P_TOT * DIMC * 2);
    f16* ph   = (f16*)alloc((size_t)P_TOT * 2048 * 2);
    f16* cath = (f16*)alloc((size_t)P_TOT * 2048 * 2);
    f16* fh   = (f16*)alloc((size_t)P_TOT * 1024 * 2);

    auto cdiv = [](size_t a, size_t b) { return (unsigned)((a + b - 1) / b); };
    dim3 T(256);

    // ---- weight prep ----
    pad_w_k<<<cdiv((size_t)1152 * 384, 256), T, 0, stream>>>(qkv_w, qkv_wh, 1152, 384, 1152, 384, 0, 0);
    pad_w_k<<<cdiv((size_t)384 * 384, 256), T, 0, stream>>>(pout_w, pout_wh, 384, 384, 384, 384, 0, 0);
    pad_w_k<<<cdiv((size_t)2048 * 384, 256), T, 0, stream>>>(pin_w, pin_wh, 2048, 384, 2042, 384, 1021, 0);
    pad_w_k<<<cdiv((size_t)1024 * 1024, 256), T, 0, stream>>>(lin_w, lin_wh, 1024, 1024, 1021, 1021, 0, 0);
    pad_w_k<<<cdiv((size_t)1024 * 2048, 256), T, 0, stream>>>(adj_w, adj_wh, 1024, 2048, 1021, 2042, 0, 1021);
    pad_w_k<<<cdiv((size_t)384 * 1024, 256), T, 0, stream>>>(fout_w, fout_wh, 384, 1024, 384, 1021, 0, 0);
    pad_b_k<<<8, T, 0, stream>>>(pin_b, pin_bp, 2048, 2042, 1021);
    pad_b_k<<<4, T, 0, stream>>>(lin_b, lin_bp, 1024, 1021, 0);
    pad_b_k<<<4, T, 0, stream>>>(adj_b, adj_bp, 1024, 1021, 0);

    // ---- attention ----
    layernorm_k<true><<<P_TOT / 8, T, 0, stream>>>(x, ln1_w, ln1_b, xn1);
    spf_norm_k<<<1, T, 0, stream>>>(spf, spfn);
    spm_k<<<cdiv((size_t)DIMC * HW, 256), T, 0, stream>>>(spfn, proj_w, proj_b, spmb);
    gemm_nt<0><<<dim3(1152 / 64, P_TOT / 128), T, 0, stream>>>(
        xn1, 384, qkv_wh, qkv_b, 384, 1152, qkvh, nullptr, nullptr, nullptr);
    dwconv_qkv_k<<<cdiv((size_t)P_TOT * 1152, 256), T, 0, stream>>>(
        qkvh, qkvdw_w, qkvdw_b, dwq, dwk, vnh);
    qnorm_k<<<8 * DIMC, T, 0, stream>>>(dwq, spmb, qsb);
    knorm_k<<<8 * DIMC, T, 0, stream>>>(dwk, krb);
    attn_k<<<64, T, 0, stream>>>(qsb, krb, temper, a1, a2, a3, a4, wmix, atM);
    outv_k<<<dim3(32, 8, 8), T, 0, stream>>>(atM, vnh, outh);
    gemm_nt<1><<<dim3(384 / 64, P_TOT / 128), T, 0, stream>>>(
        outh, 384, pout_wh, pout_b, 384, 0, nullptr, xattn, x, nullptr);

    // ---- feed-forward ----
    layernorm_k<false><<<P_TOT / 8, T, 0, stream>>>(xattn, ln2_w, ln2_b, xn2);
    gemm_nt<0><<<dim3(2048 / 64, P_TOT / 128), T, 0, stream>>>(
        xn2, 384, pin_wh, pin_bp, 384, 2048, ph, nullptr, nullptr, nullptr);
    dwconv_hid_k<<<cdiv((size_t)P_TOT * 1024, 256), T, 0, stream>>>(
        ph, dw_w, dw_b, cath);
    gemm_nt<0><<<dim3(1024 / 64, P_TOT / 128), T, 0, stream>>>(
        ph + 1024, 2048, lin_wh, lin_bp, 1024, 2048, cath + 1024,
        nullptr, nullptr, nullptr);
    gemm_nt<2><<<dim3(1024 / 64, P_TOT / 128), T, 0, stream>>>(
        cath, 2048, adj_wh, adj_bp, 2048, 1024, fh, nullptr, nullptr, nullptr);
    gemm_nt<3><<<dim3(384 / 64, P_TOT / 128), T, 0, stream>>>(
        fh, 1024, fout_wh, fout_b, 1024, 0, nullptr, nullptr, xattn, dout);
}
